// Encoder_3667902071415
// MI455X (gfx1250) — matrix operand order repaired and checked by reading, not compiled, not run on hardware
//
#include <hip/hip_runtime.h>
#include <stdint.h>
#include <math.h>

#define H 64
#define MH 128

typedef __attribute__((ext_vector_type(16))) __bf16 v16bf;
typedef __attribute__((ext_vector_type(8)))  float  v8f;

__device__ __forceinline__ float bf2f(uint32_t u) {
  union { uint32_t i; float f; } x; x.i = u << 16; return x.f;
}
__device__ __forceinline__ unsigned short f2bf(float f) {
  union { float f; uint32_t u; } x; x.f = f;
  uint32_t u = x.u;
  uint32_t r = (u + 0x7FFFu + ((u >> 16) & 1u)) >> 16;
  return (unsigned short)r;
}

union Frag16 { v16bf v; uint4 q[2]; };

// ---------------- Fourier encoder: x[N,1] -> bf16 [N,64] (sin | cos) ----------------
__global__ void fourier_kernel(const float* __restrict__ x, unsigned short* __restrict__ out, int n) {
  int i = blockIdx.x * blockDim.x + threadIdx.x;
  if (i >= n * 32) return;
  int node = i >> 5, j = i & 31;
  float m = ldexpf(x[node], 16 - j);           // x / 2^(j-16)
  out[(size_t)node * H + j]      = f2bf(sinf(m));
  out[(size_t)node * H + j + 32] = f2bf(cosf(m));
}

// ---------------- f32 -> bf16 convert ----------------
__global__ void cvt_f2bf_kernel(const float* __restrict__ in, unsigned short* __restrict__ out, int n) {
  int i = blockIdx.x * blockDim.x + threadIdx.x;
  if (i < n) out[i] = f2bf(in[i]);
}

// ---------------- copy bf16 [N,64] into columns 64..127 of [N,128] ----------------
__global__ void copycols_kernel(const unsigned short* __restrict__ src64,
                                unsigned short* __restrict__ dst128, int n) {
  int i = blockIdx.x * blockDim.x + threadIdx.x;
  if (i >= n * H) return;
  int node = i >> 6, h = i & 63;
  dst128[(size_t)node * MH + H + h] = src64[i];
}

// ---------------- edge kernel: one wave per edge ----------------
// S[dst] += relu(li[dst] + lj[src] + eattr*we);  deg[dst] += 1
__global__ void edge_kernel(int nE, const int* __restrict__ src, const int* __restrict__ dst,
                            const float* __restrict__ eattr, const float* __restrict__ we,
                            const unsigned short* __restrict__ li, const unsigned short* __restrict__ lj,
                            float* __restrict__ S, float* __restrict__ deg) {
  int gid = blockIdx.x * blockDim.x + threadIdx.x;
  int e = gid >> 5;
  int lane = threadIdx.x & 31;
  if (e >= nE) return;
  int d = dst[e], s = src[e];
  float a = eattr[e];
  int h = lane * 2;
  uint32_t pli = *(const uint32_t*)(li + (size_t)d * H + h);
  uint32_t plj = *(const uint32_t*)(lj + (size_t)s * H + h);
  float v0 = bf2f(pli & 0xffffu) + bf2f(plj & 0xffffu) + a * we[h];
  float v1 = bf2f(pli >> 16)     + bf2f(plj >> 16)     + a * we[h + 1];
  v0 = fmaxf(v0, 0.f); v1 = fmaxf(v1, 0.f);
  atomicAdd(S + (size_t)d * H + h,     v0);
  atomicAdd(S + (size_t)d * H + h + 1, v1);
  if (lane == 0) atomicAdd(deg + d, 1.0f);
}

// ---------------- WMMA GEMM: out = epilogue(A[N,K](bf16,lda) @ W[O,K]^T) ----------------
// epilogue: acc += bias[col]; acc += rowv[row]*rowb[col];
//           acc = (bng? bng[col]*bnmul : 1)*acc + (bnb? bnb[col]:0); optional relu; store bf16.
__global__ void gemm_bf16_kernel(const unsigned short* __restrict__ A, int lda,
                                 const unsigned short* __restrict__ W,
                                 int N, int K, int O,
                                 const float* __restrict__ bias,
                                 const float* __restrict__ rowv,
                                 const float* __restrict__ rowb,
                                 const float* __restrict__ bng, float bnmul,
                                 const float* __restrict__ bnb,
                                 int relu,
                                 unsigned short* __restrict__ outb, int ldo, int coloff) {
  int gid  = blockIdx.x * blockDim.x + threadIdx.x;
  int wave = gid >> 5;
  int lane = threadIdx.x & 31;
  int ctiles = O >> 4;
  int rtiles = (N + 15) >> 4;
  if (wave >= rtiles * ctiles) return;
  int row_base = (wave / ctiles) << 4;
  int o_base   = (wave % ctiles) << 4;

  int half = lane >> 4;
  int l15  = lane & 15;

  int arow = row_base + l15; if (arow > N - 1) arow = N - 1;
  const unsigned short* ap = A + (size_t)arow * lda;
  const unsigned short* wp = W + (size_t)(o_base + l15) * K + half * 16;

  v8f c = {0.f, 0.f, 0.f, 0.f, 0.f, 0.f, 0.f, 0.f};
  for (int k = 0; k < K; k += 32) {
    Frag16 af, bf_;
    af.q[0]  = *(const uint4*)(ap + k + 16 * half);       // K = k+16h .. +7
    af.q[1]  = *(const uint4*)(ap + k + 16 * half + 8);   // K = k+16h+8 .. +15
    bf_.q[0] = *(const uint4*)(wp + k);                   // K = k+16h .. +7
    bf_.q[1] = *(const uint4*)(wp + k + 8);               // K = k+16h+8 .. +15
    c = __builtin_amdgcn_wmma_f32_16x16x32_bf16(false, af.v, false, bf_.v,
                                                (short)0, c, false, false);
  }

  int col = o_base + l15;
  float biasv = bias ? bias[col] : 0.f;
  float rbv   = rowb ? rowb[col] : 0.f;
  float gg    = bng  ? bng[col] * bnmul : 1.f;
  float bb    = bnb  ? bnb[col] : 0.f;
  for (int r = 0; r < 8; ++r) {
    int row = row_base + r + 8 * half;
    if (row >= N) break;
    float acc = c[r] + biasv;
    if (rowv) acc += rowv[row] * rbv;
    acc = gg * acc + bb;
    if (relu) acc = fmaxf(acc, 0.f);
    outb[(size_t)row * ldo + coloff + col] = f2bf(acc);
  }
}

// ---------------- head final: out = h[N,128] . w2[128] + b2 ----------------
__global__ void head_final_kernel(const unsigned short* __restrict__ h,
                                  const float* __restrict__ w2, const float* __restrict__ b2,
                                  float* __restrict__ o1, float* __restrict__ o2, int n) {
  int i = blockIdx.x * blockDim.x + threadIdx.x;
  if (i >= n) return;
  float acc = b2[0];
  const uint32_t* row = (const uint32_t*)(h + (size_t)i * MH);
  for (int k = 0; k < MH / 2; ++k) {
    uint32_t u = row[k];
    acc += bf2f(u & 0xffffu) * w2[2 * k] + bf2f(u >> 16) * w2[2 * k + 1];
  }
  o1[i] = acc;
  if (o2) o2[i] = acc;
}

extern "C" void kernel_launch(void* const* d_in, const int* in_sizes, int n_in,
                              void* d_out, int out_size, void* d_ws, size_t ws_size,
                              hipStream_t stream) {
  (void)n_in; (void)out_size; (void)ws_size;
  const float* x_s   = (const float*)d_in[0];
  const float* x_t   = (const float*)d_in[1];
  const int*   ei    = (const int*)d_in[2];
  const float* eattr = (const float*)d_in[3];
  const int NS = in_sizes[0];
  const int NT = in_sizes[1];
  const int E  = in_sizes[3];
  const int NMAX = NS > NT ? NS : NT;

  char* wsb = (char*)d_ws;
  size_t off = 0;
  auto alloc = [&](size_t bytes) -> char* {
    char* p = wsb + off;
    off = (off + bytes + 255) & ~(size_t)255;
    return p;
  };

  unsigned short* xsA = (unsigned short*)alloc((size_t)NS * H * 2);
  unsigned short* xtA = (unsigned short*)alloc((size_t)NT * H * 2);
  unsigned short* xsB = (unsigned short*)alloc((size_t)NS * H * 2);
  unsigned short* xtB = (unsigned short*)alloc((size_t)NT * H * 2);
  unsigned short* li  = (unsigned short*)alloc((size_t)NMAX * H * 2);
  unsigned short* lj  = (unsigned short*)alloc((size_t)NMAX * H * 2);
  float*          S   = (float*)alloc((size_t)NMAX * H * 4);
  unsigned short* Sbf = (unsigned short*)alloc((size_t)NMAX * H * 2);
  float*          deg = (float*)alloc((size_t)NMAX * 4);
  unsigned short* cat = (unsigned short*)alloc((size_t)NMAX * MH * 2);
  unsigned short* h1  = (unsigned short*)alloc((size_t)NMAX * H * 2);

  // conv slots: 0,1 = conv_s_t[0..1]; 2,3 = conv_t_s[0..1]
  const int convBase[4] = {4, 14, 24, 34};
  // heads in dict order: xs_mu, xt_mu, xs_ls, xt_ls
  const int headBase[4] = {44, 50, 56, 62};

  auto cvt = [&](const void* srcf, int n) -> unsigned short* {
    unsigned short* dst = (unsigned short*)alloc((size_t)n * 2);
    cvt_f2bf_kernel<<<(n + 255) / 256, 256, 0, stream>>>((const float*)srcf, dst, n);
    return dst;
  };

  unsigned short *Wl_bf[4], *Wr_bf[4], *Wf_bf[4], *Wo1_bf[4], *Wo2_bf[4], *W1h_bf[4];
  for (int c = 0; c < 4; ++c) {
    int b = convBase[c];
    Wl_bf[c]  = cvt(d_in[b + 0], H * H);
    Wr_bf[c]  = cvt(d_in[b + 3], H * H);
    Wf_bf[c]  = cvt(d_in[b + 4], H * H);
    Wo1_bf[c] = cvt(d_in[b + 6], H * 2 * H);
    Wo2_bf[c] = cvt(d_in[b + 8], H * H);
  }
  for (int hd = 0; hd < 4; ++hd) W1h_bf[hd] = cvt(d_in[headBase[hd]], MH * H);

  auto gemm = [&](const unsigned short* A, int lda, const unsigned short* W,
                  int N, int K, int O,
                  const float* bias, const float* rowv, const float* rowb,
                  const float* bng, float bnmul, const float* bnb, int relu,
                  unsigned short* outb, int ldo, int coloff) {
    int tiles   = ((N + 15) / 16) * (O / 16);
    int threads = tiles * 32;
    int blocks  = (threads + 255) / 256;
    gemm_bf16_kernel<<<blocks, 256, 0, stream>>>(A, lda, W, N, K, O, bias, rowv, rowb,
                                                 bng, bnmul, bnb, relu, outb, ldo, coloff);
  };

  // Fourier encode both sides
  fourier_kernel<<<(NS * 32 + 255) / 256, 256, 0, stream>>>(x_s, xsA, NS);
  fourier_kernel<<<(NT * 32 + 255) / 256, 256, 0, stream>>>(x_t, xtA, NT);

  // One bipartite conv (Wf hoisted past the scatter-sum)
  auto bgconv = [&](int slot, const unsigned short* left_bf, int n_left,
                    const unsigned short* right_bf, int n_right,
                    const int* srcIdx, const int* dstIdx, unsigned short* out_bf) {
    int b = convBase[slot];
    const float* bl  = (const float*)d_in[b + 1];
    const float* we  = (const float*)d_in[b + 2];
    const float* bfv = (const float*)d_in[b + 5];
    const float* bo1 = (const float*)d_in[b + 7];
    const float* bo2 = (const float*)d_in[b + 9];

    // li = right @ Wl^T + bl ; lj = left @ Wr^T   (stored bf16 for edge gathers)
    gemm(right_bf, H, Wl_bf[slot], n_right, H, H, bl, 0, 0, 0, 0.f, 0, 0, li, H, 0);
    gemm(left_bf,  H, Wr_bf[slot], n_left,  H, H, 0,  0, 0, 0, 0.f, 0, 0, lj, H, 0);

    hipMemsetAsync(S,   0, (size_t)n_right * H * 4, stream);
    hipMemsetAsync(deg, 0, (size_t)n_right * 4,     stream);

    // S[dst] += relu(li[dst] + lj[src] + e*we); deg[dst] += 1
    edge_kernel<<<((size_t)E * 32 + 255) / 256, 256, 0, stream>>>(E, srcIdx, dstIdx, eattr,
                                                                  we, li, lj, S, deg);

    // agg = S @ Wf^T + deg ⊗ bf  -> cat[:, 0:64]
    cvt_f2bf_kernel<<<(n_right * H + 255) / 256, 256, 0, stream>>>(S, Sbf, n_right * H);
    gemm(Sbf, H, Wf_bf[slot], n_right, H, H, 0, deg, bfv, 0, 0.f, 0, 0, cat, MH, 0);

    // cat[:, 64:128] = right
    copycols_kernel<<<(n_right * H + 255) / 256, 256, 0, stream>>>(right_bf, cat, n_right);

    // h1 = relu(cat @ Wo1^T + bo1); out = h1 @ Wo2^T + bo2
    gemm(cat, MH, Wo1_bf[slot], n_right, MH, H, bo1, 0, 0, 0, 0.f, 0, 1, h1, H, 0);
    gemm(h1,  H,  Wo2_bf[slot], n_right, H,  H, bo2, 0, 0, 0, 0.f, 0, 0, out_bf, H, 0);
  };

  unsigned short *xs_cur = xsA, *xt_cur = xtA, *xs_new = xsB, *xt_new = xtB;
  for (int L = 0; L < 2; ++L) {
    // s -> t : left=xs, right=xt, src=row0, dst=row1
    bgconv(L,     xs_cur, NS, xt_cur, NT, ei,     ei + E, xt_new);
    // t -> s : left=xt, right=xs, src=row1, dst=row0
    bgconv(2 + L, xt_cur, NT, xs_cur, NS, ei + E, ei,     xs_new);
    unsigned short* t;
    t = xs_cur; xs_cur = xs_new; xs_new = t;
    t = xt_cur; xt_cur = xt_new; xt_new = t;
  }

  // Heads: Linear(64->128) + BN(eval) + ReLU fused into GEMM epilogue, then dot to scalar
  const float rs = 1.0f / sqrtf(1.0f + 1e-5f);
  float* out = (float*)d_out;
  auto head = [&](int hslot, const unsigned short* x, int n, float* o1, float* o2) {
    int b = headBase[hslot];
    gemm(x, H, W1h_bf[hslot], n, H, MH,
         (const float*)d_in[b + 1],          // b1
         0, 0,
         (const float*)d_in[b + 2], rs,      // g * rsqrt(1+eps)
         (const float*)d_in[b + 3],          // b (shift)
         1, cat, MH, 0);
    head_final_kernel<<<(n + 255) / 256, 256, 0, stream>>>(cat,
        (const float*)d_in[b + 4], (const float*)d_in[b + 5], o1, o2, n);
  };

  // output tuple: (xs_mu, xs_ls, xs_mu, xt_mu, xt_ls, xt_mu) flattened
  head(0, xs_cur, NS, out,                          out + 2 * (size_t)NS);
  head(2, xs_cur, NS, out + NS,                     nullptr);
  head(1, xt_cur, NT, out + 3 * (size_t)NS,                      out + 3 * (size_t)NS + 2 * (size_t)NT);
  head(3, xt_cur, NT, out + 3 * (size_t)NS + (size_t)NT,         nullptr);
}
